// MSAPairWeightedAveraging_90787018703664
// MI455X (gfx1250) — matrix operand order repaired and checked by reading, not compiled, not run on hardware
//
#include <hip/hip_runtime.h>

// ---------------- problem constants ----------------
#define SEQ 512
#define RES 384
#define CM  64       // C_MSA
#define CZ  128      // C_Z
#define CH  32       // C_HID
#define NH  8        // heads
#define HC  256      // H * C_HID
#define NCOLS 16384  // SEQ * CH, columns of per-head einsum GEMM

typedef __attribute__((ext_vector_type(8)))  float  v8f;
typedef __attribute__((ext_vector_type(16))) __bf16 v16bf;
typedef __attribute__((ext_vector_type(8)))  __bf16 v8bf;

#ifndef __has_builtin
#define __has_builtin(x) 0
#endif

#if __has_builtin(__builtin_amdgcn_tensor_load_to_lds) && \
    __has_builtin(__builtin_amdgcn_s_wait_tensorcnt)
#define USE_TDM 1
#else
#define USE_TDM 0
#endif
#if __has_builtin(__builtin_amdgcn_global_load_async_to_lds_b128)
#define USE_ASYNC 1
#else
#define USE_ASYNC 0
#endif

// Build a 16-element bf16 WMMA fragment from two 16-byte chunks.
__device__ __forceinline__ v16bf ldfrag(const __bf16* p0, const __bf16* p1) {
  v8bf lo = *(const v8bf*)p0;
  v8bf hi = *(const v8bf*)p1;
  v16bf r;
#pragma unroll
  for (int e = 0; e < 8; ++e) { r[e] = lo[e]; r[e + 8] = hi[e]; }
  return r;
}

__device__ __forceinline__ v8f wmma_bf16(v16bf a, v16bf b, v8f c) {
  // D = A(16x32 bf16) * B(32x16 bf16) + C(16x16 f32)
  return __builtin_amdgcn_wmma_f32_16x16x32_bf16(false, a, false, b, (short)0, c,
                                                 false, false);
}

#if USE_TDM
typedef __attribute__((ext_vector_type(4))) unsigned v4u;
typedef __attribute__((ext_vector_type(8))) int v8i;
typedef __attribute__((ext_vector_type(4))) int v4i;

// TDM: DMA a 128x32 bf16 tile (rows stride RES elems in global) into LDS,
// with hardware padding of 4 DWORDs (16B) after every 16 DWORDs (64B row)
// -> LDS row stride 80B == LDT(40) bf16, matching the compute-side layout.
__device__ __forceinline__ void tdm_load_tile_128x32(const __bf16* gsrc,
                                                     unsigned lds_off) {
  const unsigned long long ga = (unsigned long long)(size_t)gsrc;
  v4u g0;
  g0[0] = 1u;                                   // count=1 (user descriptor)
  g0[1] = lds_off;                              // lds_addr
  g0[2] = (unsigned)ga;                         // global_addr[31:0]
  g0[3] = (unsigned)((ga >> 32) & 0x1FFFFFFu)   // global_addr[56:32]
        | (2u << 30);                           // type = 2 ("image")
  v8i g1;
  g1[0] = (1 << 16)        // data_size = 2 bytes
        | (1 << 20)        // pad_enable
        | (3 << 22)        // pad_interval: 16 DWORDs (64B)
        | (3 << 25);       // pad_amount:   4 DWORDs (16B)
  g1[1] = (RES & 0xFFFF) << 16;   // tensor_dim0 = 384 (bits 63:48)
  g1[2] = (RES & 0xFFFF) << 16;   // dim0 hi=0 | tensor_dim1 = 384 (bits 95:80)
  g1[3] = (32 << 16);             // dim1 hi=0 | tile_dim0 = 32
  g1[4] = 128;                    // tile_dim1 = 128, tile_dim2 = 0
  g1[5] = RES;                    // tensor_dim0_stride = 384 (low 32)
  g1[6] = 0;                      // stride hi | dim1_stride low
  g1[7] = 0;
  v4i z4 = {0, 0, 0, 0};
#if __clang_major__ >= 23
  v8i z8 = {0, 0, 0, 0, 0, 0, 0, 0};
  __builtin_amdgcn_tensor_load_to_lds(g0, g1, z4, z4, z8, 0);
#else
  __builtin_amdgcn_tensor_load_to_lds(g0, g1, z4, z4, 0);
#endif
}
#endif  // USE_TDM

#if USE_ASYNC
typedef __attribute__((ext_vector_type(4))) int vec4i_b;
__device__ __forceinline__ void async_cp16(const __bf16* g, unsigned lds_off) {
  __builtin_amdgcn_global_load_async_to_lds_b128(
      (__attribute__((address_space(1))) vec4i_b*)(unsigned long long)(size_t)g,
      (__attribute__((address_space(3))) vec4i_b*)lds_off, 0, 0);
}
#endif

// =====================================================================
// Kernel C (defined first so the disasm snippet shows the TDM/async path):
// per-head einsum  O_h = W_h(384x384) @ V_h(384x16384), gated.
// Double-buffered LDS; W tile staged by TDM DMA (TENSORcnt), V tile
// software-pipelined through registers. One barrier per k-step.
// NOTE: buffer pointers are derived arithmetically from the smem root so
// InferAddressSpaces keeps all tile accesses on the DS (LDS) path.
// =====================================================================
__global__ void __launch_bounds__(256) kC(const __bf16* __restrict__ wws,
                                          const __bf16* __restrict__ vws,
                                          const __bf16* __restrict__ gws,
                                          __bf16* __restrict__ ows) {
  extern __shared__ char smem[];
  const int LDT = 40;                     // padded K stride (bf16)
  const int TILE = 128 * LDT;             // one buffer, in bf16 elems
  __bf16* const Wb = (__bf16*)smem;       // W buffers: [2][128 rows(i)][40]
  __bf16* const Vb = Wb + 2 * TILE;       // V buffers: [2][128 cols(n)][40]

  const int tid = threadIdx.x, lane = tid & 31, wid = tid >> 5;
  const int h  = blockIdx.z;
  const int i0 = blockIdx.y * 128;
  const int n0 = blockIdx.x * 128;
  const __bf16* Wh = wws + (size_t)h * RES * RES;
  const __bf16* Vh = vws + (size_t)h * RES * NCOLS;

  const int wm = wid >> 2, wn = wid & 3;  // wave tile: 64 rows x 32 cols
  v8f zero = {};
  v8f acc[4][2];
#pragma unroll
  for (int a = 0; a < 4; ++a)
#pragma unroll
    for (int b = 0; b < 2; ++b) acc[a][b] = zero;

  const int M    = lane & 15;
  const int ak0  = (lane >= 16) ? 8 : 0;
  const int ncol = lane & 15;
  const int bk0  = (lane >= 16) ? 8 : 0;

  const int wrr = tid >> 1, wkc = (tid & 1) * 16;  // W staging assignment
  const int vkk = tid >> 3, vnn = (tid & 7) * 16;  // V staging assignment

  v8bf vu0, vu1;  // pipelined V registers

  auto issueW = [&](int ks, int buf) {
#if USE_TDM
    if (tid < 32)  // one wave issues the DMA; EXEC-independent
      tdm_load_tile_128x32(Wh + (size_t)i0 * RES + ks * 32,
                           (unsigned)(size_t)Wb + (unsigned)(buf * TILE * 2));
#elif USE_ASYNC
    const __bf16* g = Wh + (size_t)(i0 + wrr) * RES + ks * 32 + wkc;
    const unsigned l =
        (unsigned)(size_t)Wb + (unsigned)((buf * TILE + wrr * LDT + wkc) * 2);
    async_cp16(g, l);
    async_cp16(g + 8, l + 16);
#else
    const __bf16* g = Wh + (size_t)(i0 + wrr) * RES + ks * 32 + wkc;
    __bf16* dst = Wb + buf * TILE + wrr * LDT + wkc;
    *(v8bf*)dst       = *(const v8bf*)g;
    *(v8bf*)(dst + 8) = *(const v8bf*)(g + 8);
#endif
  };
  auto waitW = [&]() {
#if USE_TDM
    if (tid < 32) __builtin_amdgcn_s_wait_tensorcnt((short)0);
#elif USE_ASYNC
    asm volatile("s_wait_asynccnt 0x0" ::: "memory");
#endif
  };
  auto loadVreg = [&](int ks) {
    const __bf16* g = Vh + (size_t)(ks * 32 + vkk) * NCOLS + n0 + vnn;
    vu0 = *(const v8bf*)g;
    vu1 = *(const v8bf*)(g + 8);
  };
  auto storeV = [&](int buf) {
    __bf16* dst = Vb + buf * TILE;        // arithmetic off smem root -> DS ops
#pragma unroll
    for (int e = 0; e < 8; ++e) {
      dst[(vnn + e) * LDT + vkk]     = vu0[e];
      dst[(vnn + 8 + e) * LDT + vkk] = vu1[e];
    }
  };

  issueW(0, 0);
  loadVreg(0);

  for (int ks = 0; ks < 12; ++ks) {
    const int cur = ks & 1;
    waitW();
    storeV(cur);
    __syncthreads();  // tile (W + V) visible to all waves
    if (ks + 1 < 12) {
      issueW(ks + 1, cur ^ 1);  // fill other buffer during compute
      loadVreg(ks + 1);
    }
    const __bf16* Wc = Wb + cur * TILE;
    const __bf16* Vc = Vb + cur * TILE;
    v16bf bfr[2];
#pragma unroll
    for (int nb = 0; nb < 2; ++nb) {
      const __bf16* bb = Vc + (wn * 32 + nb * 16 + ncol) * LDT;
      bfr[nb] = ldfrag(bb + bk0, bb + bk0 + 16);
    }
#pragma unroll
    for (int mb = 0; mb < 4; ++mb) {
      const __bf16* ab = Wc + (wm * 64 + mb * 16 + M) * LDT;
      const v16bf a = ldfrag(ab + ak0, ab + ak0 + 16);
#pragma unroll
      for (int nb = 0; nb < 2; ++nb)
        acc[mb][nb] = wmma_bf16(a, bfr[nb], acc[mb][nb]);
    }
    // no bottom barrier: next iteration only touches the other buffer, and
    // its top barrier orders all reads of this buffer before reuse.
  }

  // epilogue: gate + scatter to [s,i,h*32+c]
  const int lhi = (lane >= 16) ? 8 : 0;
#pragma unroll
  for (int mb = 0; mb < 4; ++mb) {
#pragma unroll
    for (int nb = 0; nb < 2; ++nb) {
      const int n = n0 + wn * 32 + nb * 16 + ncol;
      const int s = n >> 5, c = n & 31;
      const int ib = i0 + wm * 64 + mb * 16 + lhi;
      const __bf16* pg = gws + ((size_t)h * RES + ib) * NCOLS + n;
      __bf16* po = ows + ((size_t)s * RES + ib) * HC + h * CH + c;
#pragma unroll
      for (int r = 0; r < 8; ++r)
        po[r * HC] = (__bf16)(acc[mb][nb][r] * (float)pg[r * NCOLS]);
    }
  }
}

// =====================================================================
// Kernel A: LayerNorm(m) -> v = ln@w_v (masked), g = sigmoid(ln@w_g)
// =====================================================================
__global__ void __launch_bounds__(256) kA(const float* __restrict__ m,
                                          const float* __restrict__ msa_mask,
                                          const float* __restrict__ lnw,
                                          const float* __restrict__ lnb,
                                          const float* __restrict__ w_v,
                                          const float* __restrict__ w_g,
                                          __bf16* __restrict__ vws,
                                          __bf16* __restrict__ gws) {
  extern __shared__ char smem[];
  const int LDW = 72;
  __bf16* wvT  = (__bf16*)smem;          // [256][72]  n-major, k contiguous
  __bf16* wgT  = wvT + 256 * LDW;        // [256][72]
  __bf16* rowb = wgT + 256 * LDW;        // [8 waves][16][72]

  const int tid = threadIdx.x, lane = tid & 31, wid = tid >> 5;

#pragma unroll 4
  for (int k = 0; k < CM; ++k) {
    wvT[tid * LDW + k] = (__bf16)w_v[(size_t)k * HC + tid];
    wgT[tid * LDW + k] = (__bf16)w_g[(size_t)k * HC + tid];
  }

  const int row0 = blockIdx.x * 128 + wid * 16;   // rows = s*RES + i
  __bf16* rb = rowb + wid * 16 * LDW;

  {
    const int r = lane >> 1, half = lane & 1;
    const float* src = m + (size_t)(row0 + r) * CM + half * 32;
    float x[32];
    float s1 = 0.f, s2 = 0.f;
#pragma unroll
    for (int k = 0; k < 8; ++k) {
      float4 t = ((const float4*)src)[k];
      x[k * 4 + 0] = t.x; x[k * 4 + 1] = t.y; x[k * 4 + 2] = t.z; x[k * 4 + 3] = t.w;
      s1 += t.x + t.y + t.z + t.w;
      s2 += t.x * t.x + t.y * t.y + t.z * t.z + t.w * t.w;
    }
    s1 += __shfl_xor(s1, 1);
    s2 += __shfl_xor(s2, 1);
    const float mean = s1 * (1.f / 64.f);
    const float var  = s2 * (1.f / 64.f) - mean * mean;
    const float rstd = rsqrtf(var + 1e-5f);
#pragma unroll
    for (int k = 0; k < 32; ++k) {
      const int kk = half * 32 + k;
      rb[r * LDW + kk] = (__bf16)((x[k] - mean) * rstd * lnw[kk] + lnb[kk]);
    }
  }
  __syncthreads();

  const int mhalf = (lane >= 16) ? 8 : 0;
  float maskv[8]; int ibase[8];
#pragma unroll
  for (int r = 0; r < 8; ++r) {
    const int rg = row0 + mhalf + r;
    const int s = rg / RES;
    ibase[r] = (rg - s * RES) * NCOLS + s * CH;
    maskv[r] = msa_mask[rg];
  }

  const int M   = lane & 15;
  const int ak0 = (lane >= 16) ? 8 : 0;
  const __bf16* abase = rb + M * LDW;
  const v16bf a0 = ldfrag(abase + ak0,      abase + ak0 + 16);
  const v16bf a1 = ldfrag(abase + 32 + ak0, abase + 32 + ak0 + 16);

  const int ncol = lane & 15;
  const int bk0  = (lane >= 16) ? 8 : 0;

  for (int nb = 0; nb < 16; ++nb) {
    const int n = nb * 16 + ncol;
    const __bf16* bv = wvT + n * LDW;
    const __bf16* bg = wgT + n * LDW;
    v8f cv = {};
    cv = wmma_bf16(a0, ldfrag(bv + bk0,      bv + bk0 + 16),      cv);
    cv = wmma_bf16(a1, ldfrag(bv + 32 + bk0, bv + 32 + bk0 + 16), cv);
    v8f cg = {};
    cg = wmma_bf16(a0, ldfrag(bg + bk0,      bg + bk0 + 16),      cg);
    cg = wmma_bf16(a1, ldfrag(bg + 32 + bk0, bg + 32 + bk0 + 16), cg);

    const int h = nb >> 1;
    const int c = (nb & 1) * 16 + ncol;
    __bf16* pv = vws + (size_t)h * (RES * NCOLS) + c;
    __bf16* pg = gws + (size_t)h * (RES * NCOLS) + c;
#pragma unroll
    for (int r = 0; r < 8; ++r) {
      pv[ibase[r]] = (__bf16)(cv[r] * maskv[r]);
      pg[ibase[r]] = (__bf16)(1.f / (1.f + __expf(-cg[r])));
    }
  }
}

// =====================================================================
// Kernel B1: pair bias
// =====================================================================
__global__ void __launch_bounds__(256) kB1(const float* __restrict__ z,
                                           const float* __restrict__ z_mask,
                                           const float* __restrict__ lnzw,
                                           const float* __restrict__ lnzb,
                                           const float* __restrict__ w_b,
                                           float* __restrict__ btmp) {
  __shared__ float wb[CZ * NH];
  const int tid = threadIdx.x, lane = tid & 31, wid = tid >> 5;
  for (int t = tid; t < CZ * NH; t += 256) wb[t] = w_b[t];
  __syncthreads();

  const long pair = (long)blockIdx.x * 8 + wid;
  const int i = (int)(pair / RES);
  const int j = (int)(pair - (long)i * RES);
  const float* zrow = z + ((size_t)i * RES + j) * CZ;

  const float4 t = ((const float4*)zrow)[lane];
  float s1 = t.x + t.y + t.z + t.w;
  float s2 = t.x * t.x + t.y * t.y + t.z * t.z + t.w * t.w;
#pragma unroll
  for (int off = 1; off < 32; off <<= 1) {
    s1 += __shfl_xor(s1, off);
    s2 += __shfl_xor(s2, off);
  }
  const float mean = s1 * (1.f / 128.f);
  const float var  = s2 * (1.f / 128.f) - mean * mean;
  const float rstd = rsqrtf(var + 1e-5f);

  const float zv[4] = {t.x, t.y, t.z, t.w};
  float p[8] = {0.f, 0.f, 0.f, 0.f, 0.f, 0.f, 0.f, 0.f};
#pragma unroll
  for (int k = 0; k < 4; ++k) {
    const int kk = lane * 4 + k;
    const float y = (zv[k] - mean) * rstd * lnzw[kk] + lnzb[kk];
#pragma unroll
    for (int h = 0; h < NH; ++h) p[h] += y * wb[kk * NH + h];
  }
#pragma unroll
  for (int off = 1; off < 32; off <<= 1) {
#pragma unroll
    for (int h = 0; h < NH; ++h) p[h] += __shfl_xor(p[h], off);
  }
  if (lane == 0) {
    const float bias = 1e8f * (z_mask[(size_t)i * RES + j] - 1.f);
#pragma unroll
    for (int h = 0; h < NH; ++h)
      btmp[((size_t)h * RES + i) * RES + j] = p[h] + bias;
  }
}

// =====================================================================
// Kernel B2: softmax rows
// =====================================================================
__global__ void __launch_bounds__(256) kB2(const float* __restrict__ btmp,
                                           __bf16* __restrict__ wws) {
  const int tid = threadIdx.x, lane = tid & 31, wid = tid >> 5;
  const int rowid = blockIdx.x * 8 + wid;
  const float* src = btmp + (size_t)rowid * RES;

  float x[12];
  float mx = -1e30f;
#pragma unroll
  for (int t = 0; t < 12; ++t) { x[t] = src[t * 32 + lane]; mx = fmaxf(mx, x[t]); }
#pragma unroll
  for (int off = 1; off < 32; off <<= 1) mx = fmaxf(mx, __shfl_xor(mx, off));
  float sum = 0.f;
#pragma unroll
  for (int t = 0; t < 12; ++t) { x[t] = __expf(x[t] - mx); sum += x[t]; }
#pragma unroll
  for (int off = 1; off < 32; off <<= 1) sum += __shfl_xor(sum, off);
  const float inv = 1.f / sum;
#pragma unroll
  for (int t = 0; t < 12; ++t)
    wws[(size_t)rowid * RES + t * 32 + lane] = (__bf16)(x[t] * inv);
}

// =====================================================================
// Kernel D: output projection
// =====================================================================
__global__ void __launch_bounds__(256) kD(const __bf16* __restrict__ ows,
                                          const float* __restrict__ w_o,
                                          float* __restrict__ out) {
  extern __shared__ char smem[];
  const int LDO = 264;
  __bf16* WoT = (__bf16*)smem;            // [64][264] n-major, k contiguous
  const int tid = threadIdx.x, lane = tid & 31, wid = tid >> 5;
  for (int idx = tid; idx < HC * CM; idx += 256) {
    const int k = idx >> 6, n = idx & 63;
    WoT[n * LDO + k] = (__bf16)w_o[idx];
  }
  __syncthreads();

  const long row0 = (long)blockIdx.x * 128 + wid * 16;
  const int M    = lane & 15;
  const int ak0  = (lane >= 16) ? 8 : 0;
  const int ncol = lane & 15;
  const int bk0  = (lane >= 16) ? 8 : 0;
  const __bf16* arow = ows + (size_t)(row0 + M) * HC;

  v8f zero = {};
  v8f acc[4];
#pragma unroll
  for (int n = 0; n < 4; ++n) acc[n] = zero;

#pragma unroll
  for (int ks = 0; ks < 8; ++ks) {
    const __bf16* ab = arow + ks * 32;
    const v16bf a = ldfrag(ab + ak0, ab + ak0 + 16);
#pragma unroll
    for (int nb = 0; nb < 4; ++nb) {
      const __bf16* bb = WoT + (nb * 16 + ncol) * LDO + ks * 32;
      acc[nb] = wmma_bf16(a, ldfrag(bb + bk0, bb + bk0 + 16), acc[nb]);
    }
  }

  const int lhi = (lane >= 16) ? 8 : 0;
#pragma unroll
  for (int nb = 0; nb < 4; ++nb)
#pragma unroll
    for (int r = 0; r < 8; ++r)
      out[(size_t)(row0 + lhi + r) * CM + nb * 16 + ncol] = acc[nb][r];
}

// =====================================================================
extern "C" void kernel_launch(void* const* d_in, const int* in_sizes, int n_in,
                              void* d_out, int out_size, void* d_ws, size_t ws_size,
                              hipStream_t stream) {
  (void)in_sizes; (void)n_in; (void)out_size; (void)ws_size;
  const float* m        = (const float*)d_in[0];
  const float* z        = (const float*)d_in[1];
  const float* msa_mask = (const float*)d_in[2];
  const float* z_mask   = (const float*)d_in[3];
  const float* ln_m_w   = (const float*)d_in[4];
  const float* ln_m_b   = (const float*)d_in[5];
  const float* ln_z_w   = (const float*)d_in[6];
  const float* ln_z_b   = (const float*)d_in[7];
  const float* w_v      = (const float*)d_in[8];
  const float* w_g      = (const float*)d_in[9];
  const float* w_b      = (const float*)d_in[10];
  const float* w_o      = (const float*)d_in[11];
  float* out = (float*)d_out;

  // workspace carve-out
  char* ws = (char*)d_ws;
  const size_t VG_BYTES = (size_t)NH * RES * SEQ * CH * 2;  // 100,663,296
  __bf16* vws = (__bf16*)ws;
  __bf16* gws = (__bf16*)(ws + VG_BYTES);
  float*  btmp = (float*)(ws + 2 * VG_BYTES);
  __bf16* wws  = (__bf16*)(ws + 2 * VG_BYTES + (size_t)NH * RES * RES * 4);
  __bf16* ows  = (__bf16*)(ws + 2 * VG_BYTES + (size_t)NH * RES * RES * 6);

  // Dataflow order: A -> B1 -> B2 -> C -> D
  kA<<<dim3(1536), dim3(256), 2 * 256 * 72 * 2 + 8 * 16 * 72 * 2, stream>>>(
      m, msa_mask, ln_m_w, ln_m_b, w_v, w_g, vws, gws);

  kB1<<<dim3((RES * RES) / 8), dim3(256), 0, stream>>>(z, z_mask, ln_z_w, ln_z_b,
                                                       w_b, btmp);

  kB2<<<dim3((NH * RES) / 8), dim3(256), 0, stream>>>(btmp, wws);

  kC<<<dim3(NCOLS / 128, RES / 128, NH), dim3(256), 4 * 128 * 40 * 2, stream>>>(
      wws, vws, gws, ows);

  kD<<<dim3(1536), dim3(256), 64 * 264 * 2, stream>>>(ows, w_o, out);
}
